// AgentModel_60026462929338
// MI455X (gfx1250) — matrix operand order repaired and checked by reading, not compiled, not run on hardware
//
#include <hip/hip_runtime.h>
#include <hip/hip_bf16.h>

#define E        100
#define EP       128            // padded K for f16 WMMA (4 ksteps of 32)
#define GE       400            // 4*E gates
#define THREADS  256
#define MAXLEN   6
#define NITEMS   3
#define EPSV     1e-8f
#define CORR     0.1f
#define TERM_REG 0.05f
#define UTT_REG  0.001f
#define PROP_REG 0.05f

typedef __attribute__((ext_vector_type(16))) _Float16 v16h;
typedef __attribute__((ext_vector_type(8)))  _Float16 v8h;
typedef __attribute__((ext_vector_type(8)))  float    v8f;

// ---------------- WMMA fragment helpers (wave32, 16x16x32 f16) ----------------
// A (16xK tile in LDS, row-major, row stride EP):
//   lane 0-15: M=lane, halves 0..15 -> K = ks*32 + 0..15
//   lane 16-31: M=lane-16, halves 0..15 -> K = ks*32 + 16..31
__device__ __forceinline__ v16h load_a_lds(const _Float16* base, int kstep, int lane) {
    int m  = lane & 15;
    int k0 = kstep * 32 + ((lane < 16) ? 0 : 16);
    const _Float16* p = base + m * EP + k0;
    v8h lo = *(const v8h*)(p);
    v8h hi = *(const v8h*)(p + 8);
    v16h r;
#pragma unroll
    for (int i = 0; i < 8; ++i) { r[i] = lo[i]; r[i + 8] = hi[i]; }
    return r;
}

// B (KxN tile from global f16 weight stored [N][ldk]):
//   lane n = N column; lanes 0-15 hold K = ks*32+0..15, lanes 16-31 K = ks*32+16..31
__device__ __forceinline__ v16h load_b_glb(const _Float16* W, int ldk, int ntile,
                                           int kstep, int lane) {
    int n  = ntile * 16 + (lane & 15);
    int kb = kstep * 32 + ((lane < 16) ? 0 : 16);
    return *(const v16h*)(W + (long)n * ldk + kb);
}

__device__ __forceinline__ v8f wmma_acc(v16h a, v16h b, v8f c) {
    return __builtin_amdgcn_wmma_f32_16x16x32_f16(false, a, false, b, (short)0, c,
                                                  false, false);
}

__device__ __forceinline__ float sigf(float x) { return 1.0f / (1.0f + __expf(-x)); }

__device__ __forceinline__ unsigned hashu(unsigned x) {
    x ^= x >> 16; x *= 0x7feb352du; x ^= x >> 15; x *= 0x846ca68bu; x ^= x >> 16;
    return x;
}
__device__ __forceinline__ float u01(unsigned s) {
    return ((hashu(s) >> 8) + 0.5f) * (1.0f / 16777216.0f);
}

// ---------------- async global->LDS embedding copy (gfx1250 ASYNCcnt path) ----
// Copies 16 rows x 256B of pre-converted f16 embedding into sX (16x128 halves).
// 256 threads: thread = (row m = tid>>4, 16B chunk = tid&15), one
// global_load_async_to_lds_b128 per lane; completion via s_wait_asynccnt.
__device__ __forceinline__ void async_embed(const _Float16* __restrict__ embf16,
                                            const int* __restrict__ sTok, int S, int s,
                                            _Float16* sX, int tid) {
    int m  = tid >> 4;
    int ch = tid & 15;
    int t  = sTok[m * S + s];
    unsigned long long g =
        (unsigned long long)(size_t)(embf16 + (long)t * EP + ch * 8);
    unsigned l = (unsigned)(size_t)sX + (unsigned)(m * 256 + ch * 16);
    asm volatile("global_load_async_to_lds_b128 %0, %1, off"
                 :: "v"(l), "v"(g) : "memory");
    asm volatile("s_wait_asynccnt 0x0" ::: "memory");
}

// ---------------- LSTM building blocks ----------------
__device__ void lstm_gates(const _Float16* sX, const _Float16* sH, float* sG,
                           const _Float16* Wih, const _Float16* Whh,
                           const float* sBias, int lane, int wave) {
    for (int tile = wave; tile < 25; tile += 8) {
        if (tile + 8 < 25) {
            __builtin_prefetch(Wih + (long)(tile + 8) * 16 * EP, 0, 1);
            __builtin_prefetch(Whh + (long)(tile + 8) * 16 * EP, 0, 1);
        }
        v8f acc = {};
#pragma unroll
        for (int ks = 0; ks < 4; ++ks)
            acc = wmma_acc(load_a_lds(sX, ks, lane), load_b_glb(Wih, EP, tile, ks, lane), acc);
#pragma unroll
        for (int ks = 0; ks < 4; ++ks)
            acc = wmma_acc(load_a_lds(sH, ks, lane), load_b_glb(Whh, EP, tile, ks, lane), acc);
        int n  = tile * 16 + (lane & 15);
        int mb = (lane < 16) ? 0 : 8;
#pragma unroll
        for (int i = 0; i < 8; ++i)
            sG[(mb + i) * GE + n] = acc[i] + sBias[n];
    }
}

__device__ void lstm_update(_Float16* sH, float* sC, const float* sG, int tid) {
    for (int idx = tid; idx < 16 * E; idx += THREADS) {
        int m = idx / E, j = idx - m * E;
        float gi = sG[m * GE + j];
        float gf = sG[m * GE + E + j];
        float gg = sG[m * GE + 2 * E + j];
        float go = sG[m * GE + 3 * E + j];
        float c2 = sigf(gf) * sC[m * 104 + j] + sigf(gi) * tanhf(gg);
        float h2 = sigf(go) * tanhf(c2);
        sC[m * 104 + j] = c2;
        sH[m * EP + j]  = (_Float16)h2;
    }
}

__device__ void encode_seq(int S, const int* sTok, const _Float16* embf16,
                           const _Float16* Wih, const _Float16* Whh,
                           const float* bih, const float* bhh,
                           _Float16* sX, _Float16* sH, float* sC, float* sG,
                           float* sBias, int tid, int lane, int wave) {
    for (int i = tid; i < GE; i += THREADS) sBias[i] = bih[i] + bhh[i];
    for (int i = tid; i < 16 * EP; i += THREADS) sH[i] = (_Float16)0.0f;
    for (int i = tid; i < 16 * 104; i += THREADS) sC[i] = 0.0f;
    __syncthreads();
    for (int s = 0; s < S; ++s) {
        async_embed(embf16, sTok, S, s, sX, tid);
        __syncthreads();
        lstm_gates(sX, sH, sG, Wih, Whh, sBias, lane, wave);
        __syncthreads();
        lstm_update(sH, sC, sG, tid);
        __syncthreads();
    }
}

// ---------------- weight/embedding converters (f32 -> padded f16) ----------------
__global__ void conv_lstm_w(const float* __restrict__ src, _Float16* __restrict__ dst) {
    int i = blockIdx.x * blockDim.x + threadIdx.x;
    if (i >= GE * EP) return;
    int n = i >> 7, k = i & 127;
    dst[i] = (k < E) ? (_Float16)src[n * E + k] : (_Float16)0.0f;
}

__global__ void conv_comb_w(const float* __restrict__ src, _Float16* __restrict__ dst) {
    int i = blockIdx.x * blockDim.x + threadIdx.x;
    if (i >= 112 * 384) return;
    int n = i / 384, kc = i - n * 384;
    int s = kc >> 7, w = kc & 127;
    dst[i] = (n < E && w < E) ? (_Float16)src[n * 300 + s * 100 + w] : (_Float16)0.0f;
}

__global__ void conv_emb(const float* __restrict__ src, _Float16* __restrict__ dst,
                         int vocab) {
    int i = blockIdx.x * blockDim.x + threadIdx.x;
    if (i >= vocab * EP) return;
    int n = i >> 7, k = i & 127;
    dst[i] = (k < E) ? (_Float16)src[n * E + k] : (_Float16)0.0f;
}

__global__ void init_out(float* out, int B) {
    if (threadIdx.x == 0 && blockIdx.x == 0) {
        long o = (long)B * 10 + B + (long)B * 6 + (long)B * 3;
        out[o + 0] = 0.0f;                      // entropy_loss
        out[o + 1] = 0.0f;                      // term_matches
        out[o + 2] = 0.0f;                      // utt_matches
        out[o + 3] = (float)(MAXLEN * B);       // MAX_LEN * B
        out[o + 4] = 0.0f;                      // prop_matches
        out[o + 5] = (float)(NITEMS * B);       // NUM_ITEMS * B
    }
}

// ---------------- main fused forward kernel ----------------
__global__ __launch_bounds__(THREADS) void agent_fwd(
    const int* __restrict__ pool, const int* __restrict__ utility,
    const int* __restrict__ m_prev, const int* __restrict__ prev_prop,
    const _Float16* __restrict__ e_ctx16, const _Float16* __restrict__ e_utt16,
    const _Float16* __restrict__ e_up16,
    const float* __restrict__ ctx_bih, const float* __restrict__ ctx_bhh,
    const float* __restrict__ utt_bih, const float* __restrict__ utt_bhh,
    const float* __restrict__ prop_bih, const float* __restrict__ prop_bhh,
    const float* __restrict__ up_bih, const float* __restrict__ up_bhh,
    const float* __restrict__ comb_b, const float* __restrict__ term_W,
    const float* __restrict__ term_b, const float* __restrict__ up_h1_W,
    const float* __restrict__ up_h1_b, const float* __restrict__ pp_W,
    const float* __restrict__ pp_b,
    const _Float16* __restrict__ w_ctx_ih, const _Float16* __restrict__ w_ctx_hh,
    const _Float16* __restrict__ w_utt_ih, const _Float16* __restrict__ w_utt_hh,
    const _Float16* __restrict__ w_prop_ih, const _Float16* __restrict__ w_prop_hh,
    const _Float16* __restrict__ w_up_ih, const _Float16* __restrict__ w_up_hh,
    const _Float16* __restrict__ w_comb,
    float* __restrict__ out, int B) {
    __shared__ _Float16 sX[16 * EP];
    __shared__ _Float16 sH[16 * EP];
    __shared__ _Float16 sCH[16 * EP], sMH[16 * EP], sPH[16 * EP], sHT[16 * EP];
    __shared__ float sC[16 * 104];
    __shared__ float sG[16 * GE];
    __shared__ float sBias[GE];
    __shared__ int sTokC[16 * 6], sTokM[16 * 6], sTokP[16 * 3], sLast[16];
    __shared__ float sLog[160];
    __shared__ float sEnt;
    __shared__ int sTermM, sUttM, sPropM;

    const int tid  = threadIdx.x;
    const int lane = tid & 31;
    const int wave = tid >> 5;

    __builtin_amdgcn_s_wait_tensorcnt(0);   // TDM counter quiesced (gfx1250)

    const long o_nodes = 0;
    const long o_aterm = (long)B * 10;
    const long o_utt   = o_aterm + B;
    const long o_prop  = o_utt + (long)B * 6;
    const long o_scal  = o_prop + (long)B * 3;

    if (tid == 0) { sEnt = 0.0f; sTermM = 0; sUttM = 0; sPropM = 0; }

    // gather tokens for this 16-row batch tile
    for (int i = tid; i < 16 * 6; i += THREADS) {
        int m = i / 6, s = i - m * 6;
        int b = blockIdx.x * 16 + m;
        int tc = 0, tm = 0;
        if (b < B) {
            tc = (s < 3) ? pool[b * 3 + s] : utility[b * 3 + (s - 3)];
            tm = m_prev[b * 6 + s];
        }
        sTokC[i] = tc; sTokM[i] = tm;
    }
    for (int i = tid; i < 16 * 3; i += THREADS) {
        int m = i / 3, s = i - m * 3;
        int b = blockIdx.x * 16 + m;
        sTokP[i] = (b < B) ? prev_prop[b * 3 + s] : 0;
    }
    __syncthreads();

    // ---- three LSTM encoders (all WMMA; embeds via async global->LDS) ----
    encode_seq(6, sTokC, e_ctx16, w_ctx_ih, w_ctx_hh, ctx_bih, ctx_bhh,
               sX, sH, sC, sG, sBias, tid, lane, wave);
    for (int i = tid; i < 16 * EP; i += THREADS) sCH[i] = sH[i];
    __syncthreads();

    encode_seq(6, sTokM, e_utt16, w_utt_ih, w_utt_hh, utt_bih, utt_bhh,
               sX, sH, sC, sG, sBias, tid, lane, wave);
    for (int i = tid; i < 16 * EP; i += THREADS) sMH[i] = sH[i];
    __syncthreads();

    encode_seq(3, sTokP, e_ctx16, w_prop_ih, w_prop_hh, prop_bih, prop_bhh,
               sX, sH, sC, sG, sBias, tid, lane, wave);
    for (int i = tid; i < 16 * EP; i += THREADS) sPH[i] = sH[i];
    for (int i = tid; i < 16 * EP; i += THREADS) sHT[i] = (_Float16)0.0f;
    __syncthreads();

    // ---- combiner: h_t = relu([c_h|m_h|p_h] @ comb_W.T + b), K=384 via WMMA ----
    for (int tile = wave; tile < 7; tile += 8) {
        v8f acc = {};
#pragma unroll
        for (int ks = 0; ks < 12; ++ks) {
            const _Float16* src = (ks < 4) ? sCH : (ks < 8) ? sMH : sPH;
            acc = wmma_acc(load_a_lds(src, ks & 3, lane),
                           load_b_glb(w_comb, 384, tile, ks, lane), acc);
        }
        int n  = tile * 16 + (lane & 15);
        int mb = (lane < 16) ? 0 : 8;
#pragma unroll
        for (int i = 0; i < 8; ++i) {
            if (n < E) {
                float v = fmaxf(acc[i] + comb_b[n], 0.0f);
                sHT[(mb + i) * EP + n] = (_Float16)v;
            }
        }
    }
    __syncthreads();

    // ---- term policy ----
    if (tid < 16) {
        int m = tid, b = blockIdx.x * 16 + m;
        float acc = term_b[0];
        for (int j = 0; j < E; ++j) acc += (float)sHT[m * EP + j] * term_W[j];
        float p = sigf(acc);
        float u = u01((unsigned)b * 2654435761u ^ 0xA341316Cu);
        float a = (u < p) ? 1.0f : 0.0f;
        float greedy = (p >= 0.5f) ? 1.0f : 0.0f;
        float g = a * p + (1.0f - a) * (1.0f - p);
        if (b < B) {
            out[o_nodes + (long)b * 10 + 0] = logf(g);
            out[o_aterm + b] = a;
            if (greedy == a) atomicAdd(&sTermM, 1);
            atomicAdd(&sEnt, TERM_REG * (p + EPSV) * logf(p + EPSV));
        }
    }
    __syncthreads();

    // ---- proposal policy (3 heads over h_t) ----
    if (tid < 48) {
        int m = tid / 3, head = tid - m * 3;
        int b = blockIdx.x * 16 + m;
        float lg[6];
        float mx = -1e30f;
        for (int k = 0; k < 6; ++k) {
            float acc = pp_b[head * 6 + k];
            const float* wrow = pp_W + (long)(head * 6 + k) * E;
            for (int j = 0; j < E; ++j) acc += (float)sHT[m * EP + j] * wrow[j];
            lg[k] = acc;
            mx = fmaxf(mx, acc);
        }
        float pr[6], sum = 0.0f;
        for (int k = 0; k < 6; ++k) { pr[k] = __expf(lg[k] - mx); sum += pr[k]; }
        float inv = 1.0f / sum;
        int greedy = 0; float best = -1.0f;
        for (int k = 0; k < 6; ++k) {
            pr[k] *= inv;
            if (pr[k] > best) { best = pr[k]; greedy = k; }
        }
        unsigned seed = (unsigned)b * 0x9E3779B9u ^ (unsigned)(head + 1) * 0x85EBCA6Bu;
        float u = u01(seed), cdf = 0.0f;
        int a = 5;
        for (int k = 0; k < 6; ++k) { cdf += pr[k]; if (u < cdf) { a = k; break; } }
        if (b < B) {
            out[o_nodes + (long)b * 10 + 7 + head] = logf(pr[a]);
            out[o_prop + (long)b * 3 + head] = (float)a;
            if (greedy == a) atomicAdd(&sPropM, 1);
            float e = 0.0f;
            for (int k = 0; k < 6; ++k) e += (pr[k] + EPSV) * logf(pr[k] + EPSV);
            atomicAdd(&sEnt, PROP_REG * e);
        }
    }
    __syncthreads();

    // ---- utterance policy: 6-step autoregressive LSTM decode (WMMA) ----
    for (int i = tid; i < 16 * EP; i += THREADS) sH[i] = sHT[i];
    for (int i = tid; i < 16 * 104; i += THREADS) sC[i] = 0.0f;
    for (int i = tid; i < GE; i += THREADS) sBias[i] = up_bih[i] + up_bhh[i];
    if (tid < 16) sLast[tid] = 0;
    __syncthreads();

    for (int s = 0; s < MAXLEN; ++s) {
        async_embed(e_up16, sLast, 1, 0, sX, tid);
        __syncthreads();
        lstm_gates(sX, sH, sG, w_up_ih, w_up_hh, sBias, lane, wave);
        __syncthreads();
        lstm_update(sH, sC, sG, tid);
        __syncthreads();

        for (int i = tid; i < 160; i += THREADS) {
            int m = i / 10, cls = i - m * 10;
            float acc = up_h1_b[cls];
            for (int j = 0; j < E; ++j) acc += (float)sH[m * EP + j] * up_h1_W[cls * E + j];
            sLog[i] = acc;
        }
        __syncthreads();

        if (tid < 16) {
            int m = tid, b = blockIdx.x * 16 + m;
            unsigned seed = (unsigned)b * 0x9E3779B9u ^ (unsigned)(100 + s) * 0x7FEB352Du;
            float lg[10], raw[10];
            for (int c = 0; c < 10; ++c) {
                lg[c]  = sLog[m * 10 + c];
                raw[c] = u01(seed + (unsigned)c * 7919u) * 2.0f - 1.0f;
            }
            float lmin = lg[0], lmax = lg[0], nmin = raw[0], nmax = raw[0];
            for (int c = 1; c < 10; ++c) {
                lmin = fminf(lmin, lg[c]); lmax = fmaxf(lmax, lg[c]);
                nmin = fminf(nmin, raw[c]); nmax = fmaxf(nmax, raw[c]);
            }
            float dn = nmax - nmin; if (dn == 0.0f) dn = 1.0f;
            float mx = -1e30f;
            for (int c = 0; c < 10; ++c) {
                float noise = lmin + (lmax - lmin) * (raw[c] - nmin) / dn;
                lg[c] = (1.0f - CORR) * lg[c] + CORR * noise;
                mx = fmaxf(mx, lg[c]);
            }
            float pr[10], sum = 0.0f;
            for (int c = 0; c < 10; ++c) { pr[c] = __expf(lg[c] - mx); sum += pr[c]; }
            float inv = 1.0f / sum;
            int greedy = 0; float best = -1.0f;
            for (int c = 0; c < 10; ++c) {
                pr[c] *= inv;
                if (pr[c] > best) { best = pr[c]; greedy = c; }
            }
            float u = u01(seed ^ 0xDEADBEEFu), cdf = 0.0f;
            int a = 9;
            for (int c = 0; c < 10; ++c) { cdf += pr[c]; if (u < cdf) { a = c; break; } }
            if (b < B) {
                out[o_nodes + (long)b * 10 + 1 + s] = logf(pr[a]);
                out[o_utt + (long)b * 6 + s] = (float)a;
                if (greedy == a) atomicAdd(&sUttM, 1);
                float e = 0.0f;
                for (int c = 0; c < 10; ++c) e += (pr[c] + EPSV) * logf(pr[c] + EPSV);
                atomicAdd(&sEnt, UTT_REG * e);
            }
            sLast[m] = a;
        }
        __syncthreads();
    }

    if (tid == 0) {
        atomicAdd(&out[o_scal + 0], sEnt);
        atomicAdd(&out[o_scal + 1], (float)sTermM);
        atomicAdd(&out[o_scal + 2], (float)sUttM);
        atomicAdd(&out[o_scal + 4], (float)sPropM);
    }
}

// ---------------- host launcher ----------------
extern "C" void kernel_launch(void* const* d_in, const int* in_sizes, int n_in,
                              void* d_out, int out_size, void* d_ws, size_t ws_size,
                              hipStream_t stream) {
    // input order: pool, utility, m_prev, prev_proposal, then params in dict order
    const int* pool      = (const int*)d_in[0];
    const int* utility   = (const int*)d_in[1];
    const int* m_prev    = (const int*)d_in[2];
    const int* prev_prop = (const int*)d_in[3];
    const float* emb_ctx  = (const float*)d_in[4];
    const float* ctx_Wih  = (const float*)d_in[5];
    const float* ctx_Whh  = (const float*)d_in[6];
    const float* ctx_bih  = (const float*)d_in[7];
    const float* ctx_bhh  = (const float*)d_in[8];
    const float* emb_utt  = (const float*)d_in[9];
    const float* utt_Wih  = (const float*)d_in[10];
    const float* utt_Whh  = (const float*)d_in[11];
    const float* utt_bih  = (const float*)d_in[12];
    const float* utt_bhh  = (const float*)d_in[13];
    const float* prop_Wih = (const float*)d_in[14];
    const float* prop_Whh = (const float*)d_in[15];
    const float* prop_bih = (const float*)d_in[16];
    const float* prop_bhh = (const float*)d_in[17];
    const float* comb_W   = (const float*)d_in[18];
    const float* comb_b   = (const float*)d_in[19];
    const float* term_W   = (const float*)d_in[20];
    const float* term_b   = (const float*)d_in[21];
    const float* up_emb   = (const float*)d_in[22];
    const float* up_Wih   = (const float*)d_in[23];
    const float* up_Whh   = (const float*)d_in[24];
    const float* up_bih   = (const float*)d_in[25];
    const float* up_bhh   = (const float*)d_in[26];
    const float* up_h1_W  = (const float*)d_in[27];
    const float* up_h1_b  = (const float*)d_in[28];
    const float* pp_W     = (const float*)d_in[29];
    const float* pp_b     = (const float*)d_in[30];

    const int B = in_sizes[0] / 3;

    // workspace: 8 padded LSTM mats (400x128 f16) + comb (112x384 f16) + 3 f16 embeds
    _Float16* w = (_Float16*)d_ws;
    const long MSZ = (long)GE * EP;  // 51200 halves
    _Float16* w_ctx_ih  = w + 0 * MSZ;
    _Float16* w_ctx_hh  = w + 1 * MSZ;
    _Float16* w_utt_ih  = w + 2 * MSZ;
    _Float16* w_utt_hh  = w + 3 * MSZ;
    _Float16* w_prop_ih = w + 4 * MSZ;
    _Float16* w_prop_hh = w + 5 * MSZ;
    _Float16* w_up_ih   = w + 6 * MSZ;
    _Float16* w_up_hh   = w + 7 * MSZ;
    _Float16* w_comb    = w + 8 * MSZ;               // 112*384 = 43008 halves
    _Float16* e_ctx16   = w_comb + 43008;            // 6*128
    _Float16* e_utt16   = e_ctx16 + 6 * EP;          // 10*128
    _Float16* e_up16    = e_utt16 + 10 * EP;         // 10*128

    const int cb = (GE * EP + 255) / 256;
    conv_lstm_w<<<cb, 256, 0, stream>>>(ctx_Wih,  w_ctx_ih);
    conv_lstm_w<<<cb, 256, 0, stream>>>(ctx_Whh,  w_ctx_hh);
    conv_lstm_w<<<cb, 256, 0, stream>>>(utt_Wih,  w_utt_ih);
    conv_lstm_w<<<cb, 256, 0, stream>>>(utt_Whh,  w_utt_hh);
    conv_lstm_w<<<cb, 256, 0, stream>>>(prop_Wih, w_prop_ih);
    conv_lstm_w<<<cb, 256, 0, stream>>>(prop_Whh, w_prop_hh);
    conv_lstm_w<<<cb, 256, 0, stream>>>(up_Wih,   w_up_ih);
    conv_lstm_w<<<cb, 256, 0, stream>>>(up_Whh,   w_up_hh);
    conv_comb_w<<<(112 * 384 + 255) / 256, 256, 0, stream>>>(comb_W, w_comb);
    conv_emb<<<(6 * EP + 255) / 256, 256, 0, stream>>>(emb_ctx, e_ctx16, 6);
    conv_emb<<<(10 * EP + 255) / 256, 256, 0, stream>>>(emb_utt, e_utt16, 10);
    conv_emb<<<(10 * EP + 255) / 256, 256, 0, stream>>>(up_emb,  e_up16, 10);

    init_out<<<1, 32, 0, stream>>>((float*)d_out, B);

    const int grid = (B + 15) / 16;
    agent_fwd<<<grid, THREADS, 0, stream>>>(
        pool, utility, m_prev, prev_prop,
        e_ctx16, e_utt16, e_up16,
        ctx_bih, ctx_bhh, utt_bih, utt_bhh, prop_bih, prop_bhh, up_bih, up_bhh,
        comb_b, term_W, term_b, up_h1_W, up_h1_b, pp_W, pp_b,
        w_ctx_ih, w_ctx_hh, w_utt_ih, w_utt_hh, w_prop_ih, w_prop_hh,
        w_up_ih, w_up_hh, w_comb,
        (float*)d_out, B);
}
